// Net_50783693308236
// MI455X (gfx1250) — compile-verified, matrix operand order checked by reading, not run on hardware
//
#include <hip/hip_runtime.h>

// ---------------------------------------------------------------------------
// MI455X / gfx1250: wave32, WMMA 16x16x32 bf16 (f32 accumulate), 320KB LDS/WGP
// Strategy: all GEMM operands live in fragment-major bf16 blobs (one 32B
// vector load per lane per k-step); f32->bf16 conversion happens exactly once
// per element, in pack passes / producers, never in WMMA inner loops.
// ---------------------------------------------------------------------------

typedef __attribute__((ext_vector_type(16))) __bf16   v16bf;
typedef __attribute__((ext_vector_type(8)))  float    v8f;
typedef __attribute__((ext_vector_type(8)))  unsigned v8u;

static __device__ __forceinline__ __bf16 f2bf(float f) {
  unsigned u = __builtin_bit_cast(unsigned, f);
  u += 0x7FFFu + ((u >> 16) & 1u);            // round-to-nearest-even
  unsigned short h = (unsigned short)(u >> 16);
  return __builtin_bit_cast(__bf16, h);
}

// Two f32 -> packed bf16x2 in one dword (low = f0, high = f1).
static __device__ __forceinline__ unsigned pkbf(float f0, float f1) {
  unsigned u0 = __builtin_bit_cast(unsigned, f0);
  unsigned u1 = __builtin_bit_cast(unsigned, f1);
  u0 += 0x7FFFu + ((u0 >> 16) & 1u);
  u1 += 0x7FFFu + ((u1 >> 16) & 1u);
  return (u1 & 0xFFFF0000u) | (u0 >> 16);
}

static __device__ __forceinline__ float fast_rcp(float x) {
#if __has_builtin(__builtin_amdgcn_rcpf)
  return __builtin_amdgcn_rcpf(x);
#else
  return 1.0f / x;
#endif
}

// Fragment-major bf16 layout: fragment (mt*KT+kt) -> 32 lanes x 16 bf16 (32B).
// Element i of lane l maps to (m = mt*16 + (l&15), k per ISA 16-bit A layout).
// apk_index is the closed-form inverse used by producers.
static __device__ __forceinline__ size_t apk_index(int m, int k, int KT) {
  const int r    = k & 31;
  const int half = (r >> 3) & 1;
  const int i    = (r & 7) + ((r >> 4) << 3);
  const int lane = half * 16 + (m & 15);
  return ((size_t)((m >> 4) * KT + (k >> 5)) * 32 + lane) * 16 + i;
}

// Gather+convert one 16x32 A fragment from row-major f32 (global or LDS).
// Per-lane data is two runs of 8 contiguous floats -> 4x float4 (b128) loads.
static __device__ __forceinline__ v16bf load_frag_a(const float* A, int lda,
                                                    int row0, int k0, int lane) {
  const int m  = row0 + (lane & 15);
  const int kh = (lane >> 4) * 8;
  const float* base = A + (size_t)m * lda + k0 + kh;
  const float4 q0 = *(const float4*)(base);
  const float4 q1 = *(const float4*)(base + 4);
  const float4 q2 = *(const float4*)(base + 16);
  const float4 q3 = *(const float4*)(base + 20);
  v8u u;
  u[0] = pkbf(q0.x, q0.y); u[1] = pkbf(q0.z, q0.w);
  u[2] = pkbf(q1.x, q1.y); u[3] = pkbf(q1.z, q1.w);
  u[4] = pkbf(q2.x, q2.y); u[5] = pkbf(q2.z, q2.w);
  u[6] = pkbf(q3.x, q3.y); u[7] = pkbf(q3.z, q3.w);
  return __builtin_bit_cast(v16bf, u);
}

// Pack an M=128 x K activation matrix into fragment-major bf16 (LDS dest).
template <int K>
static __device__ __forceinline__ void pack_a_lds(const float* A, int lda,
                                                  __bf16* Apk, int tid) {
  constexpr int KT = K / 32;
  const int lane = tid & 31;
  for (int fr = tid >> 5; fr < 8 * KT; fr += 4) {  // 4 waves
    const int mt = fr / KT, kt = fr - mt * KT;
    v16bf a = load_frag_a(A, lda, mt * 16, kt * 32, lane);
    *(v16bf*)(Apk + ((size_t)fr * 32 + lane) * 16) = a;
  }
}

// One wave computes a 16x16 f32 tile; both operands are fragment-major bf16.
//   C = act( (pre ? pre*invc : 0) + (bias ? bias : 0) + A @ W )
// PACKOUT: store result as packed bf16 A-fragments for the next GEMM stage.
template <int K, bool PACKOUT>
static __device__ __forceinline__ void wave_gemm_tile(
    const __bf16* __restrict__ Apk, const __bf16* __restrict__ Wp,
    void* OUT, int ldo,
    const float* pre, int ldp, const float* invc,
    const float* bias, bool dorelu, int tm, int tn, int lane) {
  constexpr int KT = K / 32;
  const int col   = tn * 16 + (lane & 15);
  const int rbase = tm * 16 + (lane >> 4) * 8;
  v8f c;
#pragma unroll
  for (int v = 0; v < 8; ++v) {
    float x0 = 0.f;
    if (pre)  x0 = pre[(size_t)(rbase + v) * ldp + col] * invc[rbase + v];
    if (bias) x0 += bias[col];
    c[v] = x0;
  }
  const __bf16* abase = Apk + ((size_t)tm * KT * 32 + lane) * 16;
  const __bf16* wbase = Wp + ((size_t)tn * KT * 32 + lane) * 16;
#pragma unroll
  for (int kt = 0; kt < KT; ++kt) {
    v16bf a = *(const v16bf*)(abase + (size_t)kt * 512);
    v16bf b = *(const v16bf*)(wbase + (size_t)kt * 512);
    c = __builtin_amdgcn_wmma_f32_16x16x32_bf16(false, a, false, b,
                                                (short)0, c, false, false);
  }
#pragma unroll
  for (int v = 0; v < 8; ++v) {
    float x = c[v];
    if (dorelu) x = fmaxf(x, 0.f);
    if constexpr (PACKOUT) {
      ((__bf16*)OUT)[apk_index(rbase + v, col, ldo >> 5)] = f2bf(x);
    } else {
      ((float*)OUT)[(size_t)(rbase + v) * ldo + col] = x;
    }
  }
}

// ---------------------------------------------------------------------------
// Weight prepack: f32 (K x N, row-major) -> fragment-major bf16 blobs.
// ---------------------------------------------------------------------------
__global__ __launch_bounds__(128) void pack_w_kernel(
    const float* __restrict__ W, __bf16* __restrict__ out, int K, int N) {
  const int idx  = blockIdx.x * blockDim.x + threadIdx.x;
  const int lane = idx & 31;
  const int frag = idx >> 5;
  const int ktiles = K >> 5;
  const int ntiles = N >> 4;
  if (frag >= ktiles * ntiles) return;
  const int nt = frag / ktiles;
  const int kt = frag - nt * ktiles;
  const int n  = nt * 16 + (lane & 15);
  const int kh = (lane >> 4) * 8;
  v8u u;
#pragma unroll
  for (int i = 0; i < 4; ++i)
    u[i] = pkbf(W[(size_t)(kt * 32 + kh + 2 * i) * N + n],
                W[(size_t)(kt * 32 + kh + 2 * i + 1) * N + n]);
#pragma unroll
  for (int i = 0; i < 4; ++i)
    u[4 + i] = pkbf(W[(size_t)(kt * 32 + 16 + kh + 2 * i) * N + n],
                    W[(size_t)(kt * 32 + 16 + kh + 2 * i + 1) * N + n]);
  *(v16bf*)(out + (size_t)idx * 16) = __builtin_bit_cast(v16bf, u);
}

// ---------------------------------------------------------------------------
// Per-graph GNN kernel: one workgroup (4 wave32) per graph. ~278KB LDS.
// ---------------------------------------------------------------------------
#define NTHR 128
#define NN   128
#define EE   1024
#define HH   128

__global__ __launch_bounds__(NTHR) void gnn_kernel(
    const float* __restrict__ x, const int* __restrict__ ei,
    const __bf16* __restrict__ pWl1, const float* __restrict__ bl1,
    const __bf16* __restrict__ pWr1, const float* __restrict__ p1,
    const __bf16* __restrict__ pWl2, const float* __restrict__ bl2,
    const __bf16* __restrict__ pWr2, const float* __restrict__ p2,
    const __bf16* __restrict__ pWl3, const float* __restrict__ bl3,
    const __bf16* __restrict__ pWr3, const float* __restrict__ p3,
    __bf16* __restrict__ r_pk) {
  __shared__ float Ha[NN * HH];   // current node features (compacted)
  __shared__ float Hb[NN * HH];   // GEMM output (Y, then layer output)
  __shared__ float Hc[NN * HH];   // scatter (segment-sum) accumulator
  __shared__ __align__(32) __bf16 Apk[NN * 256];  // packed A fragments
  __shared__ int   es[EE], ed[EE];
  __shared__ float ew[EE];
  __shared__ float cnt[NN], invc[NN], score[NN];
  __shared__ int   rnk[NN];
  __shared__ float rsum[2 * HH];
  __shared__ float red[4];

  const int g    = blockIdx.x;
  const int tid  = threadIdx.x;
  const int lane = tid & 31;
  const int wave = tid >> 5;

  const float* xg = x + (size_t)g * NN * 256;
  const int*   eg = ei + (size_t)g * 2 * EE;

  for (int e = tid; e < EE; e += NTHR) {
    es[e] = eg[e];
    ed[e] = eg[EE + e];
    ew[e] = 1.f;
  }
  for (int i = tid; i < 2 * HH; i += NTHR) rsum[i] = 0.f;

  const __bf16* WlA[3] = {pWl1, pWl2, pWl3};
  const __bf16* WrA[3] = {pWr1, pWr2, pWr3};
  const float*  blA[3] = {bl1, bl2, bl3};
  const float*  pA[3]  = {p1, p2, p3};

  int ncur = NN;
  for (int layer = 0; layer < 3; ++layer) {
    // ---- zero accumulators + pack A fragments (convert once per layer) ----
    for (int i = tid; i < NN; i += NTHR) cnt[i] = 0.f;
    for (int i = tid; i < NN * HH; i += NTHR) Hc[i] = 0.f;
    if (tid == 0) red[0] = 0.f;
    if (layer == 0) pack_a_lds<256>(xg, 256, Apk, tid);
    else            pack_a_lds<128>(Ha, HH, Apk, tid);
    __syncthreads();

    // ---- degree counts (LDS float atomics) + Y = A @ Wl -> Hb (WMMA) ----
    for (int e = tid; e < EE; e += NTHR)
      if (ew[e] != 0.f) atomicAdd(&cnt[ed[e]], ew[e]);
    if (layer == 0) {
      for (int t = wave; t < 64; t += 4)
        wave_gemm_tile<256, false>(Apk, WlA[0], Hb, HH, nullptr, 0, nullptr,
                                   nullptr, false, t >> 3, t & 7, lane);
    } else {
      for (int t = wave; t < 64; t += 4)
        wave_gemm_tile<128, false>(Apk, WlA[layer], Hb, HH, nullptr, 0,
                                   nullptr, nullptr, false, t >> 3, t & 7, lane);
    }
    __syncthreads();

    // ---- invc + scatter: Hc[dst] += w * Y[src] ----
    for (int i = tid; i < NN; i += NTHR) invc[i] = fast_rcp(fmaxf(cnt[i], 1.f));
    for (int e = wave; e < EE; e += 4) {
      const float w = ew[e];
      if (w != 0.f) {
        const int s = es[e], d = ed[e];
        for (int f = lane; f < HH; f += 32)
          atomicAdd(&Hc[d * HH + f], w * Hb[s * HH + f]);
      }
    }
    __syncthreads();

    // ---- O = relu(Hc*invc + bl + A @ Wr) -> Hb (WMMA) ----
    if (layer == 0) {
      for (int t = wave; t < 64; t += 4)
        wave_gemm_tile<256, false>(Apk, WrA[0], Hb, HH, Hc, HH, invc, blA[0],
                                   true, t >> 3, t & 7, lane);
    } else {
      for (int t = wave; t < 64; t += 4)
        wave_gemm_tile<128, false>(Apk, WrA[layer], Hb, HH, Hc, HH, invc,
                                   blA[layer], true, t >> 3, t & 7, lane);
    }
    __syncthreads();

    // ---- ||p||^2 ----
    if (tid < HH) {
      const float v = pA[layer][tid];
      atomicAdd(&red[0], v * v);
    }
    __syncthreads();
#if __has_builtin(__builtin_amdgcn_rsqf)
    const float rpnorm = __builtin_amdgcn_rsqf(red[0]);
#else
    const float rpnorm = rsqrtf(red[0]);
#endif

    // ---- scores ----
    if (tid < NN) {
      float sc = -3.0e38f;
      if (tid < ncur) {
        float acc = 0.f;
        for (int f = 0; f < HH; ++f) acc += Hb[tid * HH + f] * pA[layer][f];
        sc = acc * rpnorm;
      }
      score[tid] = sc;
    }
    __syncthreads();

    // ---- stable descending rank (matches jax.lax.top_k ordering) ----
    const int kkeep = (4 * ncur + 4) / 5;  // ceil(0.8 * ncur)
    if (tid < NN) {
      int r = NN;
      if (tid < ncur) {
        const float st = score[tid];
        r = 0;
        for (int j = 0; j < ncur; ++j) {
          const float sj = score[j];
          r += (sj > st) || (sj == st && j < tid);
        }
      }
      rnk[tid] = r;
    }
    __syncthreads();

    // ---- compact Hb -> Ha (gated by tanh(score)); zero dropped rows ----
    for (int t = wave; t < NN; t += 4) {
      const int r = (t < ncur) ? rnk[t] : NN;
      if (r < kkeep) {
        const float sc = tanhf(score[t]);
        for (int f = lane; f < HH; f += 32) Ha[r * HH + f] = Hb[t * HH + f] * sc;
      }
    }
    for (int rr = kkeep + wave; rr < NN; rr += 4)
      for (int f = lane; f < HH; f += 32) Ha[rr * HH + f] = 0.f;
    __syncthreads();

    // ---- edge relabel + readout (max / mean over kept rows) ----
    for (int e = tid; e < EE; e += NTHR) {
      const int s = es[e], d = ed[e];
      const int nsr = (rnk[s] < kkeep) ? rnk[s] : -1;
      const int ndr = (rnk[d] < kkeep) ? rnk[d] : -1;
      ew[e] = ew[e] * ((nsr >= 0 && ndr >= 0) ? 1.f : 0.f);
      es[e] = nsr < 0 ? 0 : nsr;
      ed[e] = ndr < 0 ? 0 : ndr;
    }
    if (tid < HH) {
      float mx = -3.0e38f, sm = 0.f;
      const float rk = fast_rcp((float)kkeep);
      for (int r = 0; r < kkeep; ++r) {
        const float v = Ha[r * HH + tid];
        mx = fmaxf(mx, v);
        sm += v;
      }
      rsum[tid]      += mx;
      rsum[HH + tid] += sm * rk;
    }
    __syncthreads();
    ncur = kkeep;
  }

  // Emit readout row directly as packed bf16 A-fragments (M=512, K=256, KT=8).
  for (int i = tid; i < 2 * HH; i += NTHR)
    r_pk[apk_index(g, i, 8)] = f2bf(rsum[i]);
}

// ---------------------------------------------------------------------------
// Dense MLP GEMM: packed bf16 A & W; ReLU; output f32 or packed bf16.
// ---------------------------------------------------------------------------
template <int K, bool PACKOUT>
__global__ __launch_bounds__(128) void gemm_mlp_kernel(
    const __bf16* __restrict__ Apk, const __bf16* __restrict__ Wp,
    const float* __restrict__ bias, void* __restrict__ OUT, int N) {
  const int lane = threadIdx.x & 31;
  const int wave = threadIdx.x >> 5;
  const int tn = blockIdx.x * 4 + wave;   // column tile
  const int tm = blockIdx.y;              // row tile
  if (tn * 16 >= N) return;               // wave-uniform: EXEC stays all-ones
  wave_gemm_tile<K, PACKOUT>(Apk, Wp, OUT, N, nullptr, 0, nullptr, bias, true,
                             tm, tn, lane);
}

// ---------------------------------------------------------------------------
extern "C" void kernel_launch(void* const* d_in, const int* in_sizes, int n_in,
                              void* d_out, int out_size, void* d_ws,
                              size_t ws_size, hipStream_t stream) {
  (void)in_sizes; (void)n_in; (void)out_size; (void)ws_size;
  const float* x   = (const float*)d_in[0];
  const int*   ei  = (const int*)d_in[1];
  const float* Wl1 = (const float*)d_in[2];
  const float* bl1 = (const float*)d_in[3];
  const float* Wr1 = (const float*)d_in[4];
  const float* p1  = (const float*)d_in[5];
  const float* Wl2 = (const float*)d_in[6];
  const float* bl2 = (const float*)d_in[7];
  const float* Wr2 = (const float*)d_in[8];
  const float* p2  = (const float*)d_in[9];
  const float* Wl3 = (const float*)d_in[10];
  const float* bl3 = (const float*)d_in[11];
  const float* Wr3 = (const float*)d_in[12];
  const float* p3  = (const float*)d_in[13];
  const float* W1  = (const float*)d_in[14];
  const float* b1  = (const float*)d_in[15];
  const float* W2  = (const float*)d_in[16];
  const float* b2  = (const float*)d_in[17];
  const float* W3  = (const float*)d_in[18];
  const float* b3  = (const float*)d_in[19];

  __bf16* pk   = (__bf16*)d_ws;
  __bf16* r_pk  = pk;                  // 512*256 activations (packed)
  __bf16* t1_pk = r_pk + 512 * 256;    // 512*128
  __bf16* t2_pk = t1_pk + 512 * 128;   // 512*64
  __bf16* pWl1 = t2_pk + 512 * 64;     // 256*128
  __bf16* pWr1 = pWl1 + 256 * 128;
  __bf16* pWl2 = pWr1 + 256 * 128;     // 128*128
  __bf16* pWr2 = pWl2 + 128 * 128;
  __bf16* pWl3 = pWr2 + 128 * 128;
  __bf16* pWr3 = pWl3 + 128 * 128;
  __bf16* pW1  = pWr3 + 128 * 128;     // 256*128
  __bf16* pW2  = pW1 + 256 * 128;      // 128*64
  __bf16* pW3  = pW2 + 128 * 64;       // 64*16384
  float* out = (float*)d_out;          // 512*16384

  // Stage 0: prepack weights (tiny; reused by 512 blocks / 32k tiles)
  pack_w_kernel<<<16, 128, 0, stream>>>(Wl1, pWl1, 256, 128);
  pack_w_kernel<<<16, 128, 0, stream>>>(Wr1, pWr1, 256, 128);
  pack_w_kernel<<<8, 128, 0, stream>>>(Wl2, pWl2, 128, 128);
  pack_w_kernel<<<8, 128, 0, stream>>>(Wr2, pWr2, 128, 128);
  pack_w_kernel<<<8, 128, 0, stream>>>(Wl3, pWl3, 128, 128);
  pack_w_kernel<<<8, 128, 0, stream>>>(Wr3, pWr3, 128, 128);
  pack_w_kernel<<<16, 128, 0, stream>>>(W1, pW1, 256, 128);
  pack_w_kernel<<<4, 128, 0, stream>>>(W2, pW2, 128, 64);
  pack_w_kernel<<<512, 128, 0, stream>>>(W3, pW3, 64, 16384);

  // Stage 1: per-graph GNN -> packed r[512][256]
  gnn_kernel<<<512, NTHR, 0, stream>>>(x, ei, pWl1, bl1, pWr1, p1, pWl2, bl2,
                                       pWr2, p2, pWl3, bl3, pWr3, p3, r_pk);
  // Stage 2: MLP (all WMMA, activations stay packed bf16 between stages)
  gemm_mlp_kernel<256, true><<<dim3(2, 32), 128, 0, stream>>>(r_pk, pW1, b1, t1_pk, 128);
  gemm_mlp_kernel<128, true><<<dim3(1, 32), 128, 0, stream>>>(t1_pk, pW2, b2, t2_pk, 64);
  gemm_mlp_kernel<64, false><<<dim3(256, 32), 128, 0, stream>>>(t2_pk, pW3, b3, out, 16384);
}
